// CGCNNSimple_2156073582919
// MI455X (gfx1250) — matrix operand order repaired and checked by reading, not compiled, not run on hardware
//
#include <hip/hip_runtime.h>
#include <math.h>

// ---------------------------------------------------------------------------
// CGCNN forward on MI455X (gfx1250), wave32 + WMMA f16->f32.
// Edge GEMM z[E,168] @ {Wi,Wu}[168,64] via v_wmma_f32_16x16x32_f16 on z tiles
// assembled in LDS (z never hits HBM). Training-mode BN handled with a
// stats pass + recompute pass: pre-activations are NEVER materialized in HBM
// (h f16 = 5 MB and packed weights live in the 192 MB L2, so re-gather is
// cheap; HBM traffic/layer ~260 MB vs ~870 MB if zW were stored).
// Workspace use: ~78 MB.
// ---------------------------------------------------------------------------

#define NATOMS 40000
#define NEDGES 640000
#define NGRAPH 256
#define AFEAT 92
#define NF 64
#define EF 40
#define FCD 128
#define NLAY 3
#define ZF 168
#define ZPAD 192
#define BN_EPS 1e-5f
#define RBF_GAMMA (39.0f/8.0f)

typedef __attribute__((ext_vector_type(16))) _Float16 v16h;
typedef __attribute__((ext_vector_type(8)))  _Float16 v8h;
typedef __attribute__((ext_vector_type(8)))  float    v8f;

union F16Frag { v16h v; v8h h[2]; };

__device__ __forceinline__ float sp_f(float x)  { return x > 20.f ? x : log1pf(__expf(x)); }
__device__ __forceinline__ float sig_f(float x) { return 1.f / (1.f + __expf(-x)); }

// --- pack Wi/Wu (fp32 [L,ZF,NF]) -> f16 transposed+padded [L][2][NF][ZPAD] ---
__global__ void cg_pack_w(const float* __restrict__ Wi, const float* __restrict__ Wu,
                          _Float16* __restrict__ wpack) {
  int idx = blockIdx.x * 256 + threadIdx.x;
  if (idx >= NLAY * 2 * NF * ZPAD) return;
  int k   = idx % ZPAD;
  int n   = (idx / ZPAD) & 63;
  int mat = (idx / (ZPAD * NF)) & 1;
  int l   = idx / (ZPAD * NF * 2);
  float v = 0.f;
  if (k < ZF) {
    const float* W = mat ? Wu : Wi;
    v = W[((size_t)l * ZF + k) * NF + n];
  }
  wpack[idx] = (_Float16)v;
}

// --- RBF expansion of bond lengths -> f16 edge features [E, EF] ---
__global__ void cg_rbf(const float* __restrict__ r, _Float16* __restrict__ e16) {
  int e = blockIdx.x * 256 + threadIdx.x;
  if (e >= NEDGES) return;
  float x = r[3 * e], y = r[3 * e + 1], z = r[3 * e + 2];
  float d = sqrtf(x * x + y * y + z * z);
  _Float16* o = e16 + (size_t)e * EF;
  #pragma unroll 8
  for (int j = 0; j < EF; ++j) {
    float t = d - (float)j * (8.0f / 39.0f);
    o[j] = (_Float16)__expf(-RBF_GAMMA * t * t);
  }
}

// --- h = atom_features @ emb_W + emb_b ; writes f32 and f16 copies ---
__global__ void cg_embed(const float* __restrict__ af, const float* __restrict__ W,
                         const float* __restrict__ b, float* __restrict__ hf32,
                         _Float16* __restrict__ hf16) {
  int idx = blockIdx.x * 256 + threadIdx.x;
  if (idx >= NATOMS * NF) return;
  int n = idx >> 6, f = idx & 63;
  float acc = b[f];
  const float* ar = af + (size_t)n * AFEAT;
  for (int k = 0; k < AFEAT; ++k) acc += ar[k] * W[k * NF + f];
  hf32[idx] = acc;
  hf16[idx] = (_Float16)acc;
}

// --- edge GEMM: per wave one 16-edge M-tile, 8 N-tiles (Wi||Wu), K=192 in 6
//     WMMA steps. SCATTER=false: accumulate BN batch stats (sum,sumsq) only.
//     SCATTER=true : apply folded BN + sigmoid*softplus from registers and
//     scatter-add messages into agg[dst] (pre-activations never stored). ---
template <bool SCATTER>
__global__ void __launch_bounds__(256)
cg_edge_gemm(const _Float16* __restrict__ hf16, const _Float16* __restrict__ e16,
             const int* __restrict__ src, const int* __restrict__ dst,
             const _Float16* __restrict__ wpack,   // this layer: [2][NF][ZPAD] f16
             const float* __restrict__ scale, const float* __restrict__ shift,  // pass2
             float* __restrict__ agg,                                           // pass2
             float* __restrict__ gSum, float* __restrict__ gSq) {               // pass1
  extern __shared__ char smem[];
  _Float16* wt = (_Float16*)smem;                     // 128 * ZPAD f16 (49152 B)
  _Float16* zt = (_Float16*)(smem + 128 * ZPAD * 2);  // 8 waves * 16 * ZPAD f16
  float* sA = (float*)(smem + 2 * 128 * ZPAD * 2);    // 128: sum | scale
  float* sB = sA + 128;                               // 128: sumsq | shift
  int*   sDst = (int*)(sB + 128);                     // 128: dst index per tile row

  const int tid = threadIdx.x;
  { // stage both weight matrices (transposed f16) into LDS: 3072 uint4
    const uint4* gsrc = (const uint4*)wpack;
    uint4* ldst = (uint4*)wt;
    for (int i = tid; i < 3072; i += 256) ldst[i] = gsrc[i];
  }
  if (tid < 128) {
    if (SCATTER) { sA[tid] = scale[tid]; sB[tid] = shift[tid]; }
    else         { sA[tid] = 0.f;        sB[tid] = 0.f; }
  }

  const int wave  = tid >> 5;
  const int lane  = tid & 31;
  const int mtile = blockIdx.x * 8 + wave;
  const int eb    = mtile * 16;

  { // build z tile [16][ZPAD]: [h_src | h_dst | e | 0-pad], 2 lanes per row
    const int r = lane >> 1, part = lane & 1;
    const int e = eb + r;
    _Float16* zr = zt + (wave * 16 + r) * ZPAD;
    if (part == 0) {
      const uint4* hs = (const uint4*)(hf16 + (size_t)src[e] * NF);
      uint4* zp = (uint4*)zr;
      #pragma unroll
      for (int i = 0; i < 8; ++i) zp[i] = hs[i];
    } else {
      const int d = dst[e];
      sDst[wave * 16 + r] = d;
      const uint4* hd = (const uint4*)(hf16 + (size_t)d * NF);
      uint4* zp = (uint4*)(zr + NF);
      #pragma unroll
      for (int i = 0; i < 8; ++i) zp[i] = hd[i];
    }
    const unsigned* ep = (const unsigned*)(e16 + (size_t)e * EF);
    unsigned* zq = (unsigned*)(zr + 2 * NF);
    #pragma unroll
    for (int i = 0; i < 10; ++i) zq[part * 10 + i] = ep[part * 10 + i];
    unsigned* zz = (unsigned*)(zr + ZF);
    #pragma unroll
    for (int i = 0; i < 6; ++i) zz[part * 6 + i] = 0u;
  }
  __syncthreads();

  const int c16  = lane & 15;
  const int half = lane >> 4;

  v8f acc[8] = {};
  #pragma unroll
  for (int ks = 0; ks < 6; ++ks) {
    // A fragment (16x32 f16): lane(half,m) holds 16 consecutive K (matches B)
    F16Frag a;
    const _Float16* ab = zt + (wave * 16 + c16) * ZPAD + ks * 32 + half * 16;
    a.h[0] = *(const v8h*)ab;
    a.h[1] = *(const v8h*)(ab + 8);
    #pragma unroll
    for (int t = 0; t < 8; ++t) {
      // B fragment (32x16): lane holds 16 consecutive K of column n (transposed W)
      const v16h b = *(const v16h*)(wt + (t * 16 + c16) * ZPAD + ks * 32 + half * 16);
      acc[t] = __builtin_amdgcn_wmma_f32_16x16x32_f16(
          false, a.v, false, b, (short)0, acc[t], false, false);
    }
  }

  if (!SCATTER) {
    // fused BN batch statistics: column sums / sumsq -> LDS -> global atomics
    #pragma unroll
    for (int t = 0; t < 8; ++t) {
      const int col = (t & 3) * 16 + c16;
      float cs = 0.f, cq = 0.f;
      #pragma unroll
      for (int v = 0; v < 8; ++v) {
        float x = acc[t][v];
        cs += x; cq += x * x;
      }
      cs += __shfl_xor(cs, 16, 32);
      cq += __shfl_xor(cq, 16, 32);
      if (half == 0) {
        const int gc = (t < 4 ? 0 : 64) + col;
        atomicAdd(&sA[gc], cs);
        atomicAdd(&sB[gc], cq);
      }
    }
    __syncthreads();
    if (tid < 128) {
      atomicAdd(&gSum[tid], sA[tid]);
      atomicAdd(&gSq[tid], sB[tid]);
    }
  } else {
    // msg = sigmoid(bn(zWi)) * softplus(bn(zWu)); scatter to agg[dst] directly
    #pragma unroll
    for (int t = 0; t < 4; ++t) {
      const int col = t * 16 + c16;
      const float scg = sA[col],      shg = sB[col];
      const float scf = sA[64 + col], shf = sB[64 + col];
      #pragma unroll
      for (int v = 0; v < 8; ++v) {
        const int rl = half * 8 + v;
        const int di = sDst[wave * 16 + rl];
        float gate = sig_f(acc[t][v]     * scg + shg);
        float filt = sp_f (acc[t + 4][v] * scf + shf);
        atomicAdd(&agg[(size_t)di * NF + col], gate * filt);
      }
    }
  }
}

// --- fold BN into scale/shift (bias cancels under batch-mean subtraction) ---
__global__ void cg_fin_edge(const float* __restrict__ gSum, const float* __restrict__ gSq,
                            const float* __restrict__ gi, const float* __restrict__ bti,
                            const float* __restrict__ gu, const float* __restrict__ btu,
                            float* __restrict__ scale, float* __restrict__ shift) {
  int t = threadIdx.x;  // 128
  float m   = gSum[t] * (1.f / NEDGES);
  float var = gSq[t] * (1.f / NEDGES) - m * m;
  float g, b;
  if (t < 64) { g = gi[t]; b = bti[t]; } else { g = gu[t - 64]; b = btu[t - 64]; }
  float sc = g * rsqrtf(var + BN_EPS);
  scale[t] = sc;
  shift[t] = b - m * sc;
}

__global__ void cg_fin_node(const float* __restrict__ gSum, const float* __restrict__ gSq,
                            const float* __restrict__ g, const float* __restrict__ b,
                            float* __restrict__ scale, float* __restrict__ shift) {
  int t = threadIdx.x;  // 64
  float m   = gSum[t] * (1.f / NATOMS);
  float var = gSq[t] * (1.f / NATOMS) - m * m;
  float sc = g[t] * rsqrtf(var + BN_EPS);
  scale[t] = sc;
  shift[t] = b[t] - m * sc;
}

// --- column sums/sumsq over [rows,64] (stride multiple of 64 keeps col fixed) ---
__global__ void cg_col_stats64(const float* __restrict__ x, int rows,
                               float* __restrict__ gSum, float* __restrict__ gSq) {
  __shared__ float sS[64], sQ[64];
  int tid = threadIdx.x;
  if (tid < 64) { sS[tid] = 0.f; sQ[tid] = 0.f; }
  __syncthreads();
  float ls = 0.f, lq = 0.f;
  int col = tid & 63;
  size_t total = (size_t)rows * 64;
  for (size_t i = (size_t)blockIdx.x * blockDim.x + tid; i < total;
       i += (size_t)gridDim.x * blockDim.x) {
    float v = x[i]; ls += v; lq += v * v;
  }
  atomicAdd(&sS[col], ls);
  atomicAdd(&sQ[col], lq);
  __syncthreads();
  if (tid < 64) { atomicAdd(&gSum[tid], sS[tid]); atomicAdd(&gSq[tid], sQ[tid]); }
}

// --- h = softplus(h + bn(agg)); refresh f16 copy for next layer's gathers ---
__global__ void cg_node_update(float* __restrict__ hf32, _Float16* __restrict__ hf16,
                               const float* __restrict__ agg, const float* __restrict__ nsc,
                               const float* __restrict__ nsh) {
  int idx = blockIdx.x * 256 + threadIdx.x;
  if (idx >= NATOMS * NF) return;
  int n = idx & 63;
  float h = sp_f(hf32[idx] + agg[idx] * nsc[n] + nsh[n]);
  hf32[idx] = h;
  hf16[idx] = (_Float16)h;
}

// --- per-graph sums + counts ---
__global__ void cg_pool(const float* __restrict__ h, const int* __restrict__ gids,
                        float* __restrict__ pSum, float* __restrict__ pCnt) {
  int idx = blockIdx.x * 256 + threadIdx.x;
  if (idx >= NATOMS * NF) return;
  int a = idx >> 6, n = idx & 63;
  int g = gids[a];
  atomicAdd(&pSum[g * NF + n], h[idx]);
  if (n == 0) atomicAdd(&pCnt[g], 1.0f);
}

// --- head: mean -> sp -> FC -> sp -> sp -> dot(out_W) + out_b ---
__global__ void cg_final(const float* __restrict__ pSum, const float* __restrict__ pCnt,
                         const float* __restrict__ fcW, const float* __restrict__ fcB,
                         const float* __restrict__ oW, const float* __restrict__ oB,
                         float* __restrict__ out) {
  __shared__ float f[NF];
  __shared__ float red[FCD];
  int g = blockIdx.x, t = threadIdx.x;  // 128 threads
  if (t < NF) {
    float c = fmaxf(pCnt[g], 1.0f);
    f[t] = sp_f(pSum[g * NF + t] / c);
  }
  __syncthreads();
  float x = fcB[t];
  #pragma unroll 8
  for (int k = 0; k < NF; ++k) x += f[k] * fcW[k * FCD + t];
  red[t] = sp_f(sp_f(x)) * oW[t];
  __syncthreads();
  for (int s = 64; s > 0; s >>= 1) {
    if (t < s) red[t] += red[t + s];
    __syncthreads();
  }
  if (t == 0) out[g] = red[0] + oB[0];
}

extern "C" void kernel_launch(void* const* d_in, const int* in_sizes, int n_in,
                              void* d_out, int out_size, void* d_ws, size_t ws_size,
                              hipStream_t stream) {
  (void)in_sizes; (void)n_in; (void)out_size; (void)ws_size;
  const float* af   = (const float*)d_in[0];
  const float* r    = (const float*)d_in[1];
  const int*   src  = (const int*)d_in[2];
  const int*   dst  = (const int*)d_in[3];
  const int*   gids = (const int*)d_in[4];
  const float* embW = (const float*)d_in[6];
  const float* embB = (const float*)d_in[7];
  const float* Wi   = (const float*)d_in[8];
  const float* gi   = (const float*)d_in[10];
  const float* bti  = (const float*)d_in[11];
  const float* Wu   = (const float*)d_in[12];
  const float* gu   = (const float*)d_in[14];
  const float* btu  = (const float*)d_in[15];
  const float* bng  = (const float*)d_in[16];
  const float* bnb  = (const float*)d_in[17];
  const float* fcW  = (const float*)d_in[18];
  const float* fcB  = (const float*)d_in[19];
  const float* oW   = (const float*)d_in[20];
  const float* oB   = (const float*)d_in[21];
  float* out = (float*)d_out;

  char* ws = (char*)d_ws;
  size_t off = 0;
  auto carve = [&](size_t bytes) -> char* {
    char* p = ws + off;
    off = (off + bytes + 255) & ~(size_t)255;
    return p;
  };
  float*    hf32  = (float*)carve((size_t)NATOMS * NF * 4);
  _Float16* hf16  = (_Float16*)carve((size_t)NATOMS * NF * 2);
  _Float16* e16   = (_Float16*)carve((size_t)NEDGES * EF * 2);
  _Float16* wpack = (_Float16*)carve((size_t)NLAY * 2 * NF * ZPAD * 2);
  float*    agg   = (float*)carve((size_t)NATOMS * NF * 4);
  float*    eS    = (float*)carve(128 * 4);
  float*    eQ    = (float*)carve(128 * 4);
  float*    eSc   = (float*)carve(128 * 4);
  float*    eSh   = (float*)carve(128 * 4);
  float*    nS    = (float*)carve(64 * 4);
  float*    nQ    = (float*)carve(64 * 4);
  float*    nSc   = (float*)carve(64 * 4);
  float*    nSh   = (float*)carve(64 * 4);
  float*    pSum  = (float*)carve((size_t)NGRAPH * NF * 4);
  float*    pCnt  = (float*)carve((size_t)NGRAPH * 4);

  cg_pack_w<<<(NLAY * 2 * NF * ZPAD + 255) / 256, 256, 0, stream>>>(Wi, Wu, wpack);
  cg_rbf<<<(NEDGES + 255) / 256, 256, 0, stream>>>(r, e16);
  cg_embed<<<(NATOMS * NF + 255) / 256, 256, 0, stream>>>(af, embW, embB, hf32, hf16);

  // wt + zt + (sum|scale,sumsq|shift) + dst-row cache
  const size_t smem = (size_t)2 * 128 * ZPAD * 2 + 2 * 128 * 4 + 128 * 4;  // 99,840 B
  for (int l = 0; l < NLAY; ++l) {
    const _Float16* wp = wpack + (size_t)l * 2 * NF * ZPAD;
    hipMemsetAsync(eS, 0, 128 * 4, stream);
    hipMemsetAsync(eQ, 0, 128 * 4, stream);
    hipMemsetAsync(nS, 0, 64 * 4, stream);
    hipMemsetAsync(nQ, 0, 64 * 4, stream);
    hipMemsetAsync(agg, 0, (size_t)NATOMS * NF * 4, stream);
    // pass 1: WMMA GEMM + BN batch statistics only (no pre-activation stores)
    cg_edge_gemm<false><<<NEDGES / 128, 256, smem, stream>>>(
        hf16, e16, src, dst, wp, nullptr, nullptr, nullptr, eS, eQ);
    cg_fin_edge<<<1, 128, 0, stream>>>(eS, eQ, gi + l * NF, bti + l * NF,
                                       gu + l * NF, btu + l * NF, eSc, eSh);
    // pass 2: recompute WMMA GEMM, apply BN + sigmoid*softplus, scatter to agg
    cg_edge_gemm<true><<<NEDGES / 128, 256, smem, stream>>>(
        hf16, e16, src, dst, wp, eSc, eSh, agg, nullptr, nullptr);
    cg_col_stats64<<<256, 256, 0, stream>>>(agg, NATOMS, nS, nQ);
    cg_fin_node<<<1, 64, 0, stream>>>(nS, nQ, bng + l * NF, bnb + l * NF, nSc, nSh);
    cg_node_update<<<(NATOMS * NF + 255) / 256, 256, 0, stream>>>(hf32, hf16, agg, nSc, nSh);
  }
  hipMemsetAsync(pSum, 0, (size_t)NGRAPH * NF * 4, stream);
  hipMemsetAsync(pCnt, 0, (size_t)NGRAPH * 4, stream);
  cg_pool<<<(NATOMS * NF + 255) / 256, 256, 0, stream>>>(hf32, gids, pSum, pCnt);
  cg_final<<<NGRAPH, FCD, 0, stream>>>(pSum, pCnt, fcW, fcB, oW, oB, out);
}
